// ConformerBlock_35639638622511
// MI455X (gfx1250) — compile-verified, matrix operand order checked by reading, not run on hardware
//
#include <hip/hip_runtime.h>
#include <hip/hip_bf16.h>
#include <stdint.h>

// ---------------------------------------------------------------------------
// Conformer block for MI455X (gfx1250, wave32, WMMA bf16 + TDM double-buffer).
// B=16 N=512 D=256 H=4 DH=64 FF=1024 IC=512 K=31 MAXP=512 SF=1
// ---------------------------------------------------------------------------

typedef __attribute__((ext_vector_type(16))) __bf16 v16bf;
typedef __attribute__((ext_vector_type(8)))  float  v8f;
typedef unsigned int u32x4 __attribute__((ext_vector_type(4)));
typedef int          i32x4 __attribute__((ext_vector_type(4)));
typedef int          i32x8 __attribute__((ext_vector_type(8)));

#define BATCH 16
#define SEQ   512
#define DIM   256
#define HEADS 4
#define DHEAD 64
#define FFDIM 1024
#define ICCH  512
#define ROWS  (BATCH * SEQ)      // 8192 token rows

#ifndef __has_builtin
#define __has_builtin(x) 0
#endif
#if __has_builtin(__builtin_amdgcn_tensor_load_to_lds) && \
    __has_builtin(__builtin_amdgcn_s_wait_tensorcnt)
#define USE_TDM 1
#else
#define USE_TDM 0
#endif

#if defined(__HIP_DEVICE_COMPILE__) && !USE_TDM
#warning "gfx1250 probe: TDM builtins NOT found -- manual LDS staging fallback in use"
#endif

// ----------------------------- bf16 helpers --------------------------------
static __device__ __forceinline__ __bf16 f2bf(float f) {
  unsigned int u = __builtin_bit_cast(unsigned int, f);
  u += 0x7FFFu + ((u >> 16) & 1u);          // round-to-nearest-even
  unsigned short h = (unsigned short)(u >> 16);
  return __builtin_bit_cast(__bf16, h);
}
static __device__ __forceinline__ __bf16 u2bf(unsigned short h) {
  return __builtin_bit_cast(__bf16, h);
}
static __device__ __forceinline__ float bf2f(__bf16 x) {
  unsigned short h = __builtin_bit_cast(unsigned short, x);
  unsigned int u = ((unsigned int)h) << 16;
  return __builtin_bit_cast(float, u);
}
static __device__ __forceinline__ float sigf(float x) {
  return 1.0f / (1.0f + __expf(-x));
}

// k-index packing for 16-bit A/B WMMA fragments (ISA 7.12.2):
// per lane: two contiguous 8-half chunks -> lowered to 2x (ds|global)_load_b128
static __device__ __forceinline__ int wk(int r, int half) {
  return ((r < 4) ? (2 * r) : (16 + 2 * (r - 4))) + 8 * half;
}
static __device__ __forceinline__ v16bf load_frag32(const __bf16* p, int half) {
  v16bf f;
#pragma unroll
  for (int r = 0; r < 8; ++r) {
    int k = wk(r, half);
    f[2 * r] = p[k];
    f[2 * r + 1] = p[k + 1];
  }
  return f;
}
static __device__ __forceinline__ void vzero(v8f& a) {
#pragma unroll
  for (int i = 0; i < 8; ++i) a[i] = 0.0f;
}

// --------------------- Tensor Data Mover 2D tile load ----------------------
static __device__ __forceinline__ void tdm_load_2d(unsigned int lds_off,
                                                   const void* gptr,
                                                   unsigned int tile_w_dw,
                                                   unsigned int tile_h,
                                                   unsigned int stride_dw) {
#if USE_TDM
  unsigned long long ga = (unsigned long long)(uintptr_t)gptr;
  u32x4 g0;
  g0.x = 1u;                                              // count=1 (valid D#)
  g0.y = lds_off;                                         // lds_addr [63:32]
  g0.z = (unsigned int)ga;                                // global_addr [95:64]
  g0.w = ((unsigned int)(ga >> 32) & 0x01FFFFFFu) | (2u << 30);  // [56:32]|type=2
  i32x8 g1;
  g1[0] = (int)(2u << 16);                                // data_size = 4 bytes
  g1[1] = (int)((tile_w_dw & 0xFFFFu) << 16);             // tensor_dim0[15:0]
  g1[2] = (int)((tile_w_dw >> 16) | ((tile_h & 0xFFFFu) << 16));
  g1[3] = (int)((tile_h >> 16) | ((tile_w_dw & 0xFFFFu) << 16)); // tile_dim0
  g1[4] = (int)(tile_h & 0xFFFFu);                        // tile_dim1 (dim2=0)
  g1[5] = (int)stride_dw;                                 // tensor_dim0_stride
  g1[6] = 0;
  g1[7] = 0;
  i32x4 z4 = {0, 0, 0, 0};
#if __clang_major__ >= 23
  i32x8 z8 = {0, 0, 0, 0, 0, 0, 0, 0};
  __builtin_amdgcn_tensor_load_to_lds(g0, g1, z4, z4, z8, 0);
#else
  __builtin_amdgcn_tensor_load_to_lds(g0, g1, z4, z4, 0);
#endif
#else
  (void)lds_off; (void)gptr; (void)tile_w_dw; (void)tile_h; (void)stride_dw;
#endif
}
// NOTE: s_wait_tensorcnt requires a compile-time constant count.
template <int CNT>
static __device__ __forceinline__ void tdm_wait() {
#if USE_TDM
  __builtin_amdgcn_s_wait_tensorcnt(CNT);
  __asm__ volatile("" ::: "memory");
#endif
}

// ----------------------------- f32 -> bf16 copy ----------------------------
__global__ __launch_bounds__(256) void f2bf_kernel(const float* __restrict__ in,
                                                   __bf16* __restrict__ out, int n) {
  int i = blockIdx.x * 256 + threadIdx.x;
  if (i < n) out[i] = f2bf(in[i]);
}

// ----------------------------- LayerNorm -----------------------------------
__global__ __launch_bounds__(256) void ln_kernel(const float* __restrict__ x,
                                                 const float* __restrict__ g,
                                                 const float* __restrict__ b,
                                                 __bf16* __restrict__ out_bf,
                                                 float* __restrict__ out_f,
                                                 int mode) {
  __shared__ float s1[256];
  __shared__ float s2[256];
  int row = blockIdx.x;
  int t = threadIdx.x;
  float v = x[(size_t)row * DIM + t];
  s1[t] = v;
  s2[t] = v * v;
  __syncthreads();
#pragma unroll
  for (int s = 128; s > 0; s >>= 1) {
    if (t < s) { s1[t] += s1[t + s]; s2[t] += s2[t + s]; }
    __syncthreads();
  }
  float mean = s1[0] * (1.0f / DIM);
  float var = s2[0] * (1.0f / DIM) - mean * mean;
  float y = (v - mean) * rsqrtf(var + 1e-5f) * g[t] + b[t];
  if (mode == 0) out_bf[(size_t)row * DIM + t] = f2bf(y);
  else           out_f [(size_t)row * DIM + t] = y;
}

// ----------------------------- Tiled WMMA GEMM ------------------------------
// C[M,N] = A[M,K](bf16,rm) @ B[K,N](bf16,rm); block tile 128x128, K-step 32,
// 8 waves (4M x 2N).  A tiles double-buffered via TDM: wave 0 issues the
// NEXT k-tile DMA before computing the current one; s_wait_tensorcnt(1)
// guarantees the current tile landed (per-wave in-order completion).
// B tile staged manually TRANSPOSED -> both fragment loads are ds_load_b128.
template <int MODE>
__global__ __launch_bounds__(256) void gemm_bf16_kernel(
    const __bf16* __restrict__ A, int lda,
    const __bf16* __restrict__ B, int ldb,
    const float* __restrict__ bias,
    const float* __restrict__ res,
    void* __restrict__ out,
    int N, int K, float alpha) {
  __shared__ __bf16 As[2][128 * 32];  // ping-pong [m][k]
  __shared__ __bf16 Bst[128 * 32];    // [n][k]  (transposed during staging)

  const int t = threadIdx.x;
  const int lane = t & 31;
  const int w = t >> 5;
  const int wm = w & 3;
  const int wn = w >> 2;
  const int m0 = blockIdx.x * 128;
  const int n0 = blockIdx.y * 128;
  const int half = lane >> 4;
  const int l16 = lane & 15;

  const unsigned int as_off0 = (unsigned int)(uintptr_t)(void*)&As[0][0];
  const unsigned int as_off1 = (unsigned int)(uintptr_t)(void*)&As[1][0];

  v8f acc[2][4];
#pragma unroll
  for (int i = 0; i < 2; ++i)
#pragma unroll
    for (int j = 0; j < 4; ++j) vzero(acc[i][j]);

  const int ksteps = K >> 5;
#if USE_TDM
  if (w == 0)  // prologue: DMA first A tile into buffer 0
    tdm_load_2d(as_off0, A + (size_t)m0 * lda, 16u, 128u,
                (unsigned int)(lda >> 1));
#endif

  for (int kt = 0; kt < ksteps; ++kt) {
    const int k0 = kt << 5;
    const int buf = kt & 1;
    // ---- stage B tile 32x128 transposed -> Bst[n][k] (all threads) ----
    {
      int row = t >> 3;               // k within tile (0..31)
      int j0 = (t & 7) * 8;           // dword offset in the 128-wide row
      const uint32_t* src = (const uint32_t*)(B + (size_t)(k0 + row) * ldb + n0);
#pragma unroll
      for (int j = 0; j < 8; ++j) {
        uint32_t v = src[j0 + j];
        int n = 2 * (j0 + j);
        Bst[n * 32 + row] = u2bf((unsigned short)(v & 0xFFFFu));
        Bst[(n + 1) * 32 + row] = u2bf((unsigned short)(v >> 16));
      }
    }
#if USE_TDM
    if (w == 0) {
      if (kt + 1 < ksteps) {  // prefetch next A tile into the other buffer
        tdm_load_2d(buf ? as_off0 : as_off1,
                    A + (size_t)m0 * lda + ((kt + 1) << 5), 16u, 128u,
                    (unsigned int)(lda >> 1));
        tdm_wait<1>();        // current tile complete, next still in flight
      } else {
        tdm_wait<0>();        // last tile: drain
      }
    }
#else
    {
      int row = t >> 1;
      int j0 = (t & 1) * 8;
      const uint32_t* src = (const uint32_t*)(A + (size_t)(m0 + row) * lda + k0);
      uint32_t* dst = (uint32_t*)&As[buf][0];
#pragma unroll
      for (int j = 0; j < 8; ++j) dst[row * 16 + j0 + j] = src[j0 + j];
    }
#endif
    __syncthreads();

    v16bf af[2], bfr[4];
#pragma unroll
    for (int sm = 0; sm < 2; ++sm)
      af[sm] = load_frag32(&As[buf][0] + (wm * 32 + sm * 16 + l16) * 32, half);
#pragma unroll
    for (int sn = 0; sn < 4; ++sn)
      bfr[sn] = load_frag32(Bst + (wn * 64 + sn * 16 + l16) * 32, half);
#pragma unroll
    for (int sm = 0; sm < 2; ++sm)
#pragma unroll
      for (int sn = 0; sn < 4; ++sn)
        acc[sm][sn] = __builtin_amdgcn_wmma_f32_16x16x32_bf16(
            false, af[sm], false, bfr[sn], (short)0, acc[sm][sn], false, false);
    __syncthreads();
  }

  // ---- epilogue ----
#pragma unroll
  for (int sm = 0; sm < 2; ++sm) {
#pragma unroll
    for (int sn = 0; sn < 4; ++sn) {
      int col = n0 + wn * 64 + sn * 16 + l16;
      float bv = bias ? bias[col] : 0.0f;
#pragma unroll
      for (int v = 0; v < 8; ++v) {
        int row = m0 + wm * 32 + sm * 16 + v + 8 * half;
        float val = acc[sm][sn][v] + bv;
        size_t idx = (size_t)row * N + col;
        if (MODE == 0) {
          ((__bf16*)out)[idx] = f2bf(val);
        } else if (MODE == 1) {
          ((__bf16*)out)[idx] = f2bf(val * sigf(val));
        } else {
          ((float*)out)[idx] = alpha * val + res[idx];
        }
      }
    }
  }
}

// ------------------------- attention scores + rel ---------------------------
__global__ __launch_bounds__(256) void attn_scores_kernel(
    const __bf16* __restrict__ qkv,   // [ROWS, 768]
    const __bf16* __restrict__ relbf, // [1025, 64]
    float* __restrict__ S) {          // [64, 512, 512]
  __shared__ float lp[8][16][32];
  const int bh = blockIdx.x;
  const int b = bh >> 2;
  const int h = bh & 3;
  const int m0 = blockIdx.y * 16;
  const int w = threadIdx.x >> 5;
  const int lane = threadIdx.x & 31;
  const int n0 = (blockIdx.z * 8 + w) * 16;
  const int half = lane >> 4;
  const int l16 = lane & 15;

  v16bf a0, a1;
  {
    const __bf16* q = qkv + ((size_t)(b * SEQ + m0 + l16)) * 768 + h * DHEAD;
    a0 = load_frag32(q, half);
    a1 = load_frag32(q + 32, half);
  }

  v8f accqk, p0, p1;
  vzero(accqk); vzero(p0); vzero(p1);

  {  // Q.K^T : B[d][m'] = K[m', d]
    const __bf16* kr = qkv + ((size_t)(b * SEQ + n0 + l16)) * 768 + 256 + h * DHEAD;
    v16bf bb = load_frag32(kr, half);
    accqk = __builtin_amdgcn_wmma_f32_16x16x32_bf16(false, a0, false, bb, (short)0, accqk, false, false);
    bb = load_frag32(kr + 32, half);
    accqk = __builtin_amdgcn_wmma_f32_16x16x32_bf16(false, a1, false, bb, (short)0, accqk, false, false);
  }

  const int u0 = m0 - n0 + 497;  // rel_emb row window base
  {
    const __bf16* rr = relbf + (size_t)(u0 + l16) * 64;
    v16bf bb = load_frag32(rr, half);
    p0 = __builtin_amdgcn_wmma_f32_16x16x32_bf16(false, a0, false, bb, (short)0, p0, false, false);
    bb = load_frag32(rr + 32, half);
    p0 = __builtin_amdgcn_wmma_f32_16x16x32_bf16(false, a1, false, bb, (short)0, p0, false, false);
  }
  {
    const __bf16* rr = relbf + (size_t)(u0 + 16 + l16) * 64;
    v16bf bb = load_frag32(rr, half);
    p1 = __builtin_amdgcn_wmma_f32_16x16x32_bf16(false, a0, false, bb, (short)0, p1, false, false);
    bb = load_frag32(rr + 32, half);
    p1 = __builtin_amdgcn_wmma_f32_16x16x32_bf16(false, a1, false, bb, (short)0, p1, false, false);
  }

#pragma unroll
  for (int v = 0; v < 8; ++v) {
    int r = v + 8 * half;
    lp[w][r][l16] = p0[v];
    lp[w][r][16 + l16] = p1[v];
  }
  __syncthreads();
  const float scale = 0.125f;
#pragma unroll
  for (int v = 0; v < 8; ++v) {
    int r = v + 8 * half;
    int c = l16;
    float rel = lp[w][r][r - c + 15];
    S[((size_t)bh * SEQ + (m0 + r)) * SEQ + n0 + c] = (accqk[v] + rel) * scale;
  }
}

// ------------------------------- softmax ------------------------------------
// reads f32 scores, writes NORMALIZED bf16 probs (feeds PV A-fragments).
__global__ __launch_bounds__(256) void softmax_kernel(const float* __restrict__ S,
                                                      __bf16* __restrict__ P) {
  __shared__ float red[256];
  const float* p = S + (size_t)blockIdx.x * SEQ;
  __bf16* q = P + (size_t)blockIdx.x * SEQ;
  int t = threadIdx.x;
  float v0 = p[t], v1 = p[t + 256];
  red[t] = fmaxf(v0, v1);
  __syncthreads();
#pragma unroll
  for (int s = 128; s > 0; s >>= 1) {
    if (t < s) red[t] = fmaxf(red[t], red[t + s]);
    __syncthreads();
  }
  float m = red[0];
  __syncthreads();
  float e0 = __expf(v0 - m), e1 = __expf(v1 - m);
  red[t] = e0 + e1;
  __syncthreads();
#pragma unroll
  for (int s = 128; s > 0; s >>= 1) {
    if (t < s) red[t] += red[t + s];
    __syncthreads();
  }
  float inv = 1.0f / red[0];
  q[t] = f2bf(e0 * inv);
  q[t + 256] = f2bf(e1 * inv);
}

// ------------------------------ P @ V ---------------------------------------
// O = probs(bf16) @ V ; V (512x64, row stride 768) staged by TDM into LDS.
__global__ __launch_bounds__(256) void attn_pv_kernel(
    const __bf16* __restrict__ P,    // [64,512,512] probs bf16
    const __bf16* __restrict__ qkv,  // [ROWS,768]
    __bf16* __restrict__ attn_out) { // [ROWS,256]
  __shared__ __bf16 Vs[SEQ * DHEAD]; // 64 KB, [k][d]
  const int bh = blockIdx.x;
  const int b = bh >> 2;
  const int h = bh & 3;
  const int t = threadIdx.x;
  const int w = t >> 5;

#if USE_TDM
  if (w == 0) {
    tdm_load_2d((unsigned int)(uintptr_t)(void*)Vs,
                qkv + ((size_t)(b * SEQ)) * 768 + 512 + h * DHEAD,
                32u /*dwords per row*/, 512u /*rows*/, 384u /*row stride dw*/);
    tdm_wait<0>();
  }
#else
  {
    uint32_t* VsU = (uint32_t*)Vs;
#pragma unroll
    for (int i = 0; i < 2; ++i) {
      int row = t * 2 + i;
      const uint32_t* src =
          (const uint32_t*)(qkv + ((size_t)(b * SEQ + row)) * 768 + 512 + h * DHEAD);
#pragma unroll
      for (int j = 0; j < 32; ++j) VsU[row * 32 + j] = src[j];
    }
  }
#endif
  __syncthreads();

  const int lane = t & 31;
  const int half = lane >> 4;
  const int l16 = lane & 15;
  const int m0 = blockIdx.y * 128 + w * 16;

  v8f acc[4];
#pragma unroll
  for (int i = 0; i < 4; ++i) vzero(acc[i]);

  for (int kt = 0; kt < 16; ++kt) {
    const int kb = kt * 32;
    const __bf16* prow = P + ((size_t)bh * SEQ + m0 + l16) * SEQ + kb;
    v16bf a = load_frag32(prow, half);
#pragma unroll
    for (int sn = 0; sn < 4; ++sn) {
      v16bf bfr;
      int n = sn * 16 + l16;
#pragma unroll
      for (int r = 0; r < 8; ++r) {
        int k = kb + wk(r, half);
        bfr[2 * r] = Vs[k * DHEAD + n];
        bfr[2 * r + 1] = Vs[(k + 1) * DHEAD + n];
      }
      acc[sn] = __builtin_amdgcn_wmma_f32_16x16x32_bf16(false, a, false, bfr, (short)0, acc[sn], false, false);
    }
  }
#pragma unroll
  for (int sn = 0; sn < 4; ++sn) {
#pragma unroll
    for (int v = 0; v < 8; ++v) {
      int row = m0 + v + 8 * half;
      int col = h * DHEAD + sn * 16 + l16;
      attn_out[(size_t)(b * SEQ + row) * DIM + col] = f2bf(acc[sn][v]);
    }
  }
}

// ----------------- GLU + depthwise conv (K=31) + BN + swish -----------------
__global__ __launch_bounds__(256) void glu_dwconv_kernel(
    const __bf16* __restrict__ pw1, const float* __restrict__ dw_w,
    const float* __restrict__ dw_b, const float* __restrict__ bn_g,
    const float* __restrict__ bn_b, const float* __restrict__ bn_m,
    const float* __restrict__ bn_v, __bf16* __restrict__ out) {
  __shared__ float gl[16][513];
  const int b = blockIdx.x;
  const int c0 = blockIdx.y * 16;
  const int t = threadIdx.x;
  const int cl = t & 15;
  const int nb = t >> 4;
  for (int i = 0; i < 32; ++i) {
    int n = i * 16 + nb;
    size_t base = ((size_t)(b * SEQ + n)) * 1024 + c0 + cl;
    float av = bf2f(pw1[base]);
    float gv = bf2f(pw1[base + 512]);
    gl[cl][n] = av * sigf(gv);
  }
  __syncthreads();
  const int c = c0 + cl;
  float wloc[31];
#pragma unroll
  for (int k = 0; k < 31; ++k) wloc[k] = dw_w[c * 31 + k];
  float rs = rsqrtf(bn_v[c] + 1e-5f);
  float sc = rs * bn_g[c];
  float off = bn_b[c] - bn_m[c] * sc;
  float db = dw_b[c];
  for (int i = 0; i < 32; ++i) {
    int n = i * 16 + nb;
    float acc = 0.0f;
#pragma unroll
    for (int k = 0; k < 31; ++k) {
      int idx = n + k - 15;
      if (idx >= 0 && idx < SEQ) acc += gl[cl][idx] * wloc[k];
    }
    float hc = (acc + db) * sc + off;
    float y = hc * sigf(hc);
    out[((size_t)(b * SEQ + n)) * ICCH + c] = f2bf(y);
  }
}

// ============================ host orchestration ============================
static void launch_gemm(const __bf16* A, int lda, const __bf16* B, int ldb,
                        const float* bias, const float* res, void* out,
                        int M, int N, int K, int mode, float alpha,
                        hipStream_t s) {
  dim3 g(M / 128, N / 128);
  if (mode == 0)
    gemm_bf16_kernel<0><<<g, 256, 0, s>>>(A, lda, B, ldb, bias, res, out, N, K, alpha);
  else if (mode == 1)
    gemm_bf16_kernel<1><<<g, 256, 0, s>>>(A, lda, B, ldb, bias, res, out, N, K, alpha);
  else
    gemm_bf16_kernel<2><<<g, 256, 0, s>>>(A, lda, B, ldb, bias, res, out, N, K, alpha);
}
static void cvt(const void* in, __bf16* out, int n, hipStream_t s) {
  f2bf_kernel<<<(n + 255) / 256, 256, 0, s>>>((const float*)in, out, n);
}

extern "C" void kernel_launch(void* const* d_in, const int* in_sizes, int n_in,
                              void* d_out, int out_size, void* d_ws, size_t ws_size,
                              hipStream_t stream) {
  (void)in_sizes; (void)n_in; (void)out_size; (void)ws_size;
  const float* x       = (const float*)d_in[0];
  const float* ff1_g   = (const float*)d_in[1];
  const float* ff1_b   = (const float*)d_in[2];
  const float* ff1_w1  = (const float*)d_in[3];
  const float* ff1_b1  = (const float*)d_in[4];
  const float* ff1_w2  = (const float*)d_in[5];
  const float* ff1_b2  = (const float*)d_in[6];
  const float* at_g    = (const float*)d_in[7];
  const float* at_b    = (const float*)d_in[8];
  const float* qkv_w   = (const float*)d_in[9];
  const float* out_w   = (const float*)d_in[10];
  const float* out_b   = (const float*)d_in[11];
  const float* rel     = (const float*)d_in[12];
  const float* cv_g    = (const float*)d_in[13];
  const float* cv_b    = (const float*)d_in[14];
  const float* pw1_w   = (const float*)d_in[15];
  const float* pw1_b   = (const float*)d_in[16];
  const float* dw_w    = (const float*)d_in[17];
  const float* dw_b    = (const float*)d_in[18];
  const float* bn_g    = (const float*)d_in[19];
  const float* bn_b    = (const float*)d_in[20];
  const float* bn_m    = (const float*)d_in[21];
  const float* bn_v    = (const float*)d_in[22];
  const float* pw2_w   = (const float*)d_in[23];
  const float* pw2_b   = (const float*)d_in[24];
  const float* ff2_g   = (const float*)d_in[25];
  const float* ff2_b   = (const float*)d_in[26];
  const float* ff2_w1  = (const float*)d_in[27];
  const float* ff2_b1  = (const float*)d_in[28];
  const float* ff2_w2  = (const float*)d_in[29];
  const float* ff2_b2  = (const float*)d_in[30];
  const float* pn_g    = (const float*)d_in[31];
  const float* pn_b    = (const float*)d_in[32];

  char* ws = (char*)d_ws;
  size_t o = 0;
  auto carve = [&](size_t bytes) { size_t r = o; o = (o + bytes + 255) & ~(size_t)255; return r; };
  float*  xbuf    = (float*)(ws + carve((size_t)ROWS * DIM * 4));
  __bf16* h_bf    = (__bf16*)(ws + carve((size_t)ROWS * DIM * 2));
  __bf16* big_bf  = (__bf16*)(ws + carve((size_t)ROWS * FFDIM * 2));
  __bf16* act2_bf = (__bf16*)(ws + carve((size_t)ROWS * ICCH * 2));
  __bf16* attn_bf = (__bf16*)(ws + carve((size_t)ROWS * DIM * 2));
  __bf16* qkv_bf  = (__bf16*)(ws + carve((size_t)ROWS * 768 * 2));
  float*  Sbuf    = (float*)(ws + carve((size_t)64 * SEQ * SEQ * 4));
  __bf16* Pbuf    = (__bf16*)(ws + carve((size_t)64 * SEQ * SEQ * 2));
  __bf16* w_ff1a  = (__bf16*)(ws + carve((size_t)DIM * FFDIM * 2));
  __bf16* w_ff1b  = (__bf16*)(ws + carve((size_t)FFDIM * DIM * 2));
  __bf16* w_qkv   = (__bf16*)(ws + carve((size_t)DIM * 768 * 2));
  __bf16* w_out   = (__bf16*)(ws + carve((size_t)DIM * DIM * 2));
  __bf16* w_rel   = (__bf16*)(ws + carve((size_t)1025 * DHEAD * 2));
  __bf16* w_pw1   = (__bf16*)(ws + carve((size_t)DIM * (2 * ICCH) * 2));
  __bf16* w_pw2   = (__bf16*)(ws + carve((size_t)ICCH * DIM * 2));
  __bf16* w_ff2a  = (__bf16*)(ws + carve((size_t)DIM * FFDIM * 2));
  __bf16* w_ff2b  = (__bf16*)(ws + carve((size_t)FFDIM * DIM * 2));

  cvt(ff1_w1, w_ff1a, DIM * FFDIM, stream);
  cvt(ff1_w2, w_ff1b, FFDIM * DIM, stream);
  cvt(qkv_w,  w_qkv,  DIM * 768,   stream);
  cvt(out_w,  w_out,  DIM * DIM,   stream);
  cvt(rel,    w_rel,  1025 * DHEAD, stream);
  cvt(pw1_w,  w_pw1,  DIM * 2 * ICCH, stream);
  cvt(pw2_w,  w_pw2,  ICCH * DIM,  stream);
  cvt(ff2_w1, w_ff2a, DIM * FFDIM, stream);
  cvt(ff2_w2, w_ff2b, FFDIM * DIM, stream);

  // ---- FF1 ----
  ln_kernel<<<ROWS, 256, 0, stream>>>(x, ff1_g, ff1_b, h_bf, nullptr, 0);
  launch_gemm(h_bf, DIM, w_ff1a, FFDIM, ff1_b1, nullptr, big_bf,
              ROWS, FFDIM, DIM, 1, 0.f, stream);
  launch_gemm(big_bf, FFDIM, w_ff1b, DIM, ff1_b2, x, xbuf,
              ROWS, DIM, FFDIM, 2, 0.5f, stream);

  // ---- relative-position attention ----
  ln_kernel<<<ROWS, 256, 0, stream>>>(xbuf, at_g, at_b, h_bf, nullptr, 0);
  launch_gemm(h_bf, DIM, w_qkv, 768, nullptr, nullptr, qkv_bf,
              ROWS, 768, DIM, 0, 0.f, stream);
  attn_scores_kernel<<<dim3(64, 32, 4), 256, 0, stream>>>(qkv_bf, w_rel, Sbuf);
  softmax_kernel<<<64 * SEQ, 256, 0, stream>>>(Sbuf, Pbuf);
  attn_pv_kernel<<<dim3(64, 4), 256, 0, stream>>>(Pbuf, qkv_bf, attn_bf);
  launch_gemm(attn_bf, DIM, w_out, DIM, out_b, xbuf, xbuf,
              ROWS, DIM, DIM, 2, 1.0f, stream);

  // ---- conformer conv module ----
  ln_kernel<<<ROWS, 256, 0, stream>>>(xbuf, cv_g, cv_b, h_bf, nullptr, 0);
  launch_gemm(h_bf, DIM, w_pw1, 2 * ICCH, pw1_b, nullptr, big_bf,
              ROWS, 2 * ICCH, DIM, 0, 0.f, stream);
  glu_dwconv_kernel<<<dim3(BATCH, ICCH / 16), 256, 0, stream>>>(
      big_bf, dw_w, dw_b, bn_g, bn_b, bn_m, bn_v, act2_bf);
  launch_gemm(act2_bf, ICCH, w_pw2, DIM, pw2_b, xbuf, xbuf,
              ROWS, DIM, ICCH, 2, 1.0f, stream);

  // ---- FF2 ----
  ln_kernel<<<ROWS, 256, 0, stream>>>(xbuf, ff2_g, ff2_b, h_bf, nullptr, 0);
  launch_gemm(h_bf, DIM, w_ff2a, FFDIM, ff2_b1, nullptr, big_bf,
              ROWS, FFDIM, DIM, 1, 0.f, stream);
  launch_gemm(big_bf, FFDIM, w_ff2b, DIM, ff2_b2, xbuf, xbuf,
              ROWS, DIM, FFDIM, 2, 0.5f, stream);

  // ---- final LayerNorm -> output (f32) ----
  ln_kernel<<<ROWS, 256, 0, stream>>>(xbuf, pn_g, pn_b, nullptr, (float*)d_out, 1);
}
